// GraphAttention_87007447482378
// MI455X (gfx1250) — matrix operand order repaired and checked by reading, not compiled, not run on hardware
//
#include <hip/hip_runtime.h>
#include <hip/hip_bf16.h>

// ---------------------------------------------------------------------------
// GAT forward for gfx1250 (MI455X): split-bf16 WMMA GEMMs + segment softmax.
// ---------------------------------------------------------------------------

typedef __attribute__((ext_vector_type(16))) __bf16 v16bf;
typedef __attribute__((ext_vector_type(8)))  __bf16 v8bf;
typedef __attribute__((ext_vector_type(8)))  float  v8f;

#define GAT_E        500000
#define GAT_N        50000
#define GAT_D        128       // in channels (K)
#define GAT_HC       128       // heads*out (N of GEMM)
#define GAT_H        4
#define GAT_C        32
#define GAT_NEG      0.01f
#define W_PITCH      136       // 128 + 8 bf16 pad (272B rows, 16B aligned)
#define MAT_ELEMS    (W_PITCH * 128)          // per hi or lo part
#define WAVES_PER_WG 16

static __device__ __forceinline__ v8f wmma_bf16(v16bf a, v16bf b, v8f c) {
  return __builtin_amdgcn_wmma_f32_16x16x32_bf16(false, a, false, b, (short)0, c,
                                                 false, false);
}

// monotone float<->uint mapping for atomicMax on signed floats
static __device__ __forceinline__ unsigned encf(float f) {
  unsigned u = __float_as_uint(f);
  return (u & 0x80000000u) ? ~u : (u | 0x80000000u);
}
static __device__ __forceinline__ float decf(unsigned u) {
  u = (u & 0x80000000u) ? (u & 0x7fffffffu) : ~u;
  return __uint_as_float(u);
}

// Warm L2/WGP$ with one 16-row (8KB) feature block: 32 lanes x 256B.
static __device__ __forceinline__ void prefetch_block(const float* __restrict__ A,
                                                      int rowBase, int lane) {
  __builtin_prefetch(A + (size_t)rowBase * GAT_D + lane * 64, 0, 3);
}

// A fragment: 16x32 bf16, lanes 0-15 row M=lane, K={0..7,16..23};
// lanes 16-31 row M=lane-16, K={8..15,24..31}. Split fp32 -> bf16 hi+lo.
static __device__ __forceinline__ void load_a_frag(const float* __restrict__ A,
                                                   int rowBase, int kBase, int lane,
                                                   v16bf& hi, v16bf& lo) {
  const int m    = lane & 15;
  const int koff = (lane & 16) ? 8 : 0;
  const float* p = A + (size_t)(rowBase + m) * GAT_D + kBase + koff;
  float4 x0 = *(const float4*)(p);
  float4 x1 = *(const float4*)(p + 4);
  float4 x2 = *(const float4*)(p + 16);
  float4 x3 = *(const float4*)(p + 20);
  float xs[16] = {x0.x, x0.y, x0.z, x0.w, x1.x, x1.y, x1.z, x1.w,
                  x2.x, x2.y, x2.z, x2.w, x3.x, x3.y, x3.z, x3.w};
#pragma unroll
  for (int i = 0; i < 16; ++i) {
    __bf16 h = (__bf16)xs[i];
    hi[i] = h;
    lo[i] = (__bf16)(xs[i] - (float)h);
  }
}

// B fragment from LDS: W^T stored [n][k] (pitch W_PITCH). Lane col n=lane&15,
// K = (lane&16 ? 8 : 0) + {0..7, 16..23}  (two 16B ds reads).
static __device__ __forceinline__ v16bf load_b_frag(const __bf16* __restrict__ Wt,
                                                    int nBase, int kBase, int lane) {
  const int n  = nBase + (lane & 15);
  const int kh = (lane & 16) ? 8 : 0;
  const __bf16* p = Wt + n * W_PITCH + kBase + kh;
  v8bf b0 = *(const v8bf*)(p);
  v8bf b1 = *(const v8bf*)(p + 16);
  return __builtin_shufflevector(b0, b1, 0, 1, 2, 3, 4, 5, 6, 7,
                                 8, 9, 10, 11, 12, 13, 14, 15);
}

// One [16 x 128] += [16 x 128] * [128 x 128] pass, split-bf16 (3 WMMA/tile).
// kt loop kept rolled (unroll 1) with double-buffered A fragment so that at
// most ~200 VGPRs are live: no spills, no vgpr-MSB bank switching.
static __device__ __forceinline__ void gemm_pass(const float* __restrict__ A, int rowBase,
                                                 const __bf16* __restrict__ Whi,
                                                 const __bf16* __restrict__ Wlo,
                                                 int lane, v8f acc[8]) {
  v16bf ahi, alo;
  load_a_frag(A, rowBase, 0, lane, ahi, alo);
#pragma unroll 1
  for (int kt = 0; kt < 4; ++kt) {
    v16bf nhi, nlo;  // prefetch next A fragment ((kt+1)&3 stays in-bounds)
    load_a_frag(A, rowBase, ((kt + 1) & 3) * 32, lane, nhi, nlo);
#pragma unroll
    for (int nt = 0; nt < 8; ++nt) {
      v16bf bhi = load_b_frag(Whi, nt * 16, kt * 32, lane);
      v16bf blo = load_b_frag(Wlo, nt * 16, kt * 32, lane);
      acc[nt] = wmma_bf16(ahi, bhi, acc[nt]);
      acc[nt] = wmma_bf16(alo, bhi, acc[nt]);
      acc[nt] = wmma_bf16(ahi, blo, acc[nt]);
    }
    ahi = nhi;
    alo = nlo;
  }
}

// Stage one [128 x 128] fp32 weight (row-major [k][n]) transposed into LDS as
// bf16 hi/lo with padded pitch.
static __device__ __forceinline__ void stage_weights(const float* __restrict__ W,
                                                     __bf16* Whi, __bf16* Wlo,
                                                     int tid, int nThreads) {
  for (int i = tid; i < GAT_D * GAT_HC; i += nThreads) {
    int k = i >> 7, n = i & 127;
    float  x = W[i];
    __bf16 h = (__bf16)x;
    Whi[n * W_PITCH + k] = h;
    Wlo[n * W_PITCH + k] = (__bf16)(x - (float)h);
  }
}

// ---------------------------------------------------------------------------
// Kernel 1: init zmax (encoded -inf == 0), denom, out1
// ---------------------------------------------------------------------------
__global__ void gat_init_kernel(unsigned* __restrict__ zmax_u, float* __restrict__ denom,
                                float* __restrict__ out1, int nzm, int nout) {
  int stride = (int)(blockDim.x * gridDim.x);
  for (int i = blockIdx.x * blockDim.x + threadIdx.x; i < nout; i += stride) {
    out1[i] = 0.0f;
    if (i < nzm) { zmax_u[i] = 0u; denom[i] = 0.0f; }
  }
}

// ---------------------------------------------------------------------------
// Kernel 2: z = sum_c leaky(s+t+e)*attn  -> zbuf[E,4]; atomicMax per sender.
// ---------------------------------------------------------------------------
__global__ __launch_bounds__(WAVES_PER_WG * 32)
void gat_z_kernel(const float* __restrict__ recvF, const float* __restrict__ sendF,
                  const float* __restrict__ edgeF, const int* __restrict__ sndIdx,
                  const float* __restrict__ Wsrc, const float* __restrict__ Wtgt,
                  const float* __restrict__ Wedge, const float* __restrict__ attn,
                  float* __restrict__ zbuf, unsigned* __restrict__ zmax_u, int nBlocks) {
  extern __shared__ __bf16 smem[];
  __bf16* WtHi = smem;
  __bf16* WtLo = WtHi + MAT_ELEMS;
  __bf16* WsHi = WtLo + MAT_ELEMS;
  __bf16* WsLo = WsHi + MAT_ELEMS;
  __bf16* WeHi = WsLo + MAT_ELEMS;
  __bf16* WeLo = WeHi + MAT_ELEMS;

  const int tid = threadIdx.x, nThr = blockDim.x;
  stage_weights(Wtgt, WtHi, WtLo, tid, nThr);
  stage_weights(Wsrc, WsHi, WsLo, tid, nThr);
  stage_weights(Wedge, WeHi, WeLo, tid, nThr);
  __syncthreads();

  const int lane = tid & 31;
  const int wave = tid >> 5;
  const int col  = lane & 15;
  const int step = (int)gridDim.x * WAVES_PER_WG;
  float at0[GAT_H], at1[GAT_H];
#pragma unroll
  for (int h = 0; h < GAT_H; ++h) {
    at0[h] = attn[h * GAT_C + col];
    at1[h] = attn[h * GAT_C + 16 + col];
  }

  for (int blk = blockIdx.x * WAVES_PER_WG + wave; blk < nBlocks; blk += step) {
    const int rowBase = blk * 16;

    // warm L2 with this wave's next grid-stride block (~1-2k cycles ahead)
    int nblk = blk + step;
    if (nblk >= nBlocks) nblk = blk;
    prefetch_block(recvF, nblk * 16, lane);
    prefetch_block(sendF, nblk * 16, lane);
    prefetch_block(edgeF, nblk * 16, lane);

    v8f acc[8];
#pragma unroll
    for (int nt = 0; nt < 8; ++nt)
#pragma unroll
      for (int r = 0; r < 8; ++r) acc[nt][r] = 0.0f;

    gemm_pass(recvF, rowBase, WtHi, WtLo, lane, acc);  // t
    gemm_pass(sendF, rowBase, WsHi, WsLo, lane, acc);  // + s
    gemm_pass(edgeF, rowBase, WeHi, WeLo, lane, acc);  // + e

    // z[m,h] = sum_c leaky(acc[m, h*32+c]) * attn[h,c]; reduce 16 lanes/half.
#pragma unroll
    for (int h = 0; h < GAT_H; ++h) {
      float zp[8];
#pragma unroll
      for (int r = 0; r < 8; ++r) {
        float v0 = acc[2 * h][r];
        v0 = (v0 > 0.0f) ? v0 : GAT_NEG * v0;
        float v1 = acc[2 * h + 1][r];
        v1 = (v1 > 0.0f) ? v1 : GAT_NEG * v1;
        zp[r] = v0 * at0[h] + v1 * at1[h];
      }
#pragma unroll
      for (int mask = 1; mask <= 8; mask <<= 1)
#pragma unroll
        for (int r = 0; r < 8; ++r) zp[r] += __shfl_xor(zp[r], mask, 32);

      if (col == h) {
        const int mOff = (lane & 16) ? 8 : 0;
#pragma unroll
        for (int r = 0; r < 8; ++r) {
          int row = rowBase + mOff + r;
          zbuf[(size_t)row * GAT_H + h] = zp[r];
          int s = sndIdx[row];
          atomicMax(&zmax_u[(size_t)s * GAT_H + h], encf(zp[r]));
        }
      }
    }
  }
}

// ---------------------------------------------------------------------------
// Kernel 3: in-place ez = exp(z - zmax[snd]); atomicAdd denom.
// ---------------------------------------------------------------------------
__global__ void gat_softmax_kernel(const int* __restrict__ sndIdx, float* __restrict__ zbuf,
                                   const unsigned* __restrict__ zmax_u,
                                   float* __restrict__ denom, int total) {
  int idx = blockIdx.x * blockDim.x + threadIdx.x;
  if (idx >= total) return;
  int e = idx >> 2, h = idx & 3;
  int s = sndIdx[e];
  float zm = decf(zmax_u[s * GAT_H + h]);
  float ez = __expf(zbuf[idx] - zm);
  zbuf[idx] = ez;
  atomicAdd(&denom[s * GAT_H + h], ez);
}

// ---------------------------------------------------------------------------
// Kernel 4: recompute t via WMMA; out2[e,c] = 0.25*sum_h t[e,h,c]*a[e,h];
//           out1[n,c] += out2 over receiver segments (atomic f32).
// ---------------------------------------------------------------------------
__global__ __launch_bounds__(WAVES_PER_WG * 32)
void gat_out_kernel(const float* __restrict__ recvF, const int* __restrict__ sndIdx,
                    const int* __restrict__ rcvIdx, const float* __restrict__ Wtgt,
                    const float* __restrict__ ezbuf, const float* __restrict__ denom,
                    float* __restrict__ out1, float* __restrict__ out2, int nBlocks) {
  extern __shared__ __bf16 smem[];
  __bf16* WtHi = smem;
  __bf16* WtLo = WtHi + MAT_ELEMS;
  stage_weights(Wtgt, WtHi, WtLo, threadIdx.x, blockDim.x);
  __syncthreads();

  const int lane = threadIdx.x & 31;
  const int wave = threadIdx.x >> 5;
  const int m    = lane & 15;
  const int step = (int)gridDim.x * WAVES_PER_WG;

  for (int blk = blockIdx.x * WAVES_PER_WG + wave; blk < nBlocks; blk += step) {
    const int rowBase = blk * 16;

    int nblk = blk + step;
    if (nblk >= nBlocks) nblk = blk;
    prefetch_block(recvF, nblk * 16, lane);

    const int myRow = rowBase + m;            // lanes 16-31 mirror lanes 0-15
    const int snd   = sndIdx[myRow];
    const int rcv   = rcvIdx[myRow];
    float aH[GAT_H];
#pragma unroll
    for (int h = 0; h < GAT_H; ++h)
      aH[h] = ezbuf[(size_t)myRow * GAT_H + h] / denom[(size_t)snd * GAT_H + h];

    v8f acc[8];
#pragma unroll
    for (int nt = 0; nt < 8; ++nt)
#pragma unroll
      for (int r = 0; r < 8; ++r) acc[nt][r] = 0.0f;

    gemm_pass(recvF, rowBase, WtHi, WtLo, lane, acc);  // t = receivers @ W_tgt

    const int mOff = (lane & 16) ? 8 : 0;
    float res[2][8];
#pragma unroll
    for (int p = 0; p < 2; ++p)
#pragma unroll
      for (int r = 0; r < 8; ++r) res[p][r] = 0.0f;

#pragma unroll
    for (int nt = 0; nt < 8; ++nt) {
      const int h = nt >> 1;
#pragma unroll
      for (int r = 0; r < 8; ++r) {
        float av = __shfl(aH[h], mOff + r, 32);  // a[row M, h] from owner lane
        res[nt & 1][r] += acc[nt][r] * av;
      }
    }
#pragma unroll
    for (int r = 0; r < 8; ++r) {
      int   rowM = rowBase + mOff + r;
      int   rv   = __shfl(rcv, mOff + r, 32);
      float v0   = 0.25f * res[0][r];
      float v1   = 0.25f * res[1][r];
      out2[(size_t)rowM * GAT_C + m]      = v0;
      out2[(size_t)rowM * GAT_C + 16 + m] = v1;
      atomicAdd(&out1[(size_t)rv * GAT_C + m],      v0);
      atomicAdd(&out1[(size_t)rv * GAT_C + 16 + m], v1);
    }
  }
}

// ---------------------------------------------------------------------------
extern "C" void kernel_launch(void* const* d_in, const int* in_sizes, int n_in,
                              void* d_out, int out_size, void* d_ws, size_t ws_size,
                              hipStream_t stream) {
  (void)in_sizes; (void)n_in; (void)out_size; (void)ws_size;
  const float* receivers = (const float*)d_in[0];
  const float* senders   = (const float*)d_in[1];
  const float* edgeAttr  = (const float*)d_in[2];
  const int*   sndIdx    = (const int*)d_in[3];
  const int*   rcvIdx    = (const int*)d_in[4];
  const float* Wsrc      = (const float*)d_in[5];
  const float* Wtgt      = (const float*)d_in[6];
  const float* Wedge     = (const float*)d_in[7];
  const float* attn      = (const float*)d_in[8];

  float* out1 = (float*)d_out;                       // [N, 32]
  float* out2 = out1 + (size_t)GAT_N * GAT_C;        // [E, 32]

  char* w = (char*)d_ws;
  float*    zbuf  = (float*)w;    w += (size_t)GAT_E * GAT_H * sizeof(float);   // 8 MB
  unsigned* zmaxU = (unsigned*)w; w += (size_t)GAT_N * GAT_H * sizeof(unsigned);
  float*    denom = (float*)w;

  const int nBlocks = GAT_E / 16;                    // 31250 (exact)
  const int smemZ   = 3 * 2 * MAT_ELEMS * (int)sizeof(__bf16);  // 208896 B
  const int smemO   = 2 * MAT_ELEMS * (int)sizeof(__bf16);      //  69632 B

  gat_init_kernel<<<(GAT_N * GAT_C + 255) / 256, 256, 0, stream>>>(
      zmaxU, denom, out1, GAT_N * GAT_H, GAT_N * GAT_C);

  gat_z_kernel<<<256, WAVES_PER_WG * 32, smemZ, stream>>>(
      receivers, senders, edgeAttr, sndIdx, Wsrc, Wtgt, Wedge, attn,
      zbuf, zmaxU, nBlocks);

  gat_softmax_kernel<<<(GAT_E * GAT_H + 255) / 256, 256, 0, stream>>>(
      sndIdx, zbuf, zmaxU, denom, GAT_E * GAT_H);

  gat_out_kernel<<<512, WAVES_PER_WG * 32, smemO, stream>>>(
      receivers, sndIdx, rcvIdx, Wtgt, zbuf, denom, out1, out2, nBlocks);
}
